// AxialMultiHeadAttention_69243462746787
// MI455X (gfx1250) — compile-verified, matrix operand order checked by reading, not run on hardware
//
#include <hip/hip_runtime.h>
#include <hip/hip_bf16.h>

typedef __bf16 bf16;
typedef __bf16 v16bf  __attribute__((ext_vector_type(16)));
typedef __bf16 bf16x2 __attribute__((ext_vector_type(2)));
typedef float  v8f    __attribute__((ext_vector_type(8)));

#define BN_EPS 1e-5f

// Load a 16x32 bf16 WMMA operand fragment from LDS.
// `base` points at a block of 16 rows, each `stride` bf16 wide, K contiguous.
// ISA 16-bit layout: lanes 0-15 = rows 0..15 holding K {0..7,16..23},
// lanes 16-31 = same rows holding K {8..15,24..31}; 2 bf16 per VGPR.
__device__ __forceinline__ v16bf load_frag16(const bf16* __restrict__ base, int stride) {
  const int lane = threadIdx.x & 31;
  const int row  = lane & 15;
  const int kb   = (lane < 16) ? 0 : 8;
  const bf16* p = base + row * stride + kb;
  v16bf a;
#pragma unroll
  for (int v = 0; v < 4; ++v) {
    bf16x2 lo = *(const bf16x2*)(p + 2 * v);
    bf16x2 hi = *(const bf16x2*)(p + 16 + 2 * v);
    a[2 * v]     = lo[0];
    a[2 * v + 1] = lo[1];
    a[8 + 2 * v]     = hi[0];
    a[8 + 2 * v + 1] = hi[1];
  }
  return a;
}

#define WMMA_BF16(A, B, C) \
  __builtin_amdgcn_wmma_f32_16x16x32_bf16(false, (A), false, (B), (short)0, (C), false, false)

// ---------------------------------------------------------------------------
// Kernel 1: qkv[b= n*64+w][o][h] = BN1( sum_c W[o][c] * x[n][c][h][w] ), bf16.
// One big GEMM per image n: C(512 x 4096) = W(512x256) x X(256x4096).
// Block tile 128x128, 8 waves each owning a 64x32 region (4x2 wmma tiles).
// ---------------------------------------------------------------------------
__global__ __launch_bounds__(256)
void qkv_gemm_bn(const float* __restrict__ x, const float* __restrict__ w,
                 const float* __restrict__ g1, const float* __restrict__ b1,
                 const float* __restrict__ m1, const float* __restrict__ v1,
                 bf16* __restrict__ qkv) {
  __shared__ bf16 Wl[128 * 34];   // A tile: [row o][k], stride 34
  __shared__ bf16 Xl[128 * 34];   // B tile transposed: [col][k], stride 34
  __shared__ float sc[128];
  __shared__ float sh[128];

  const int tid  = threadIdx.x;
  const int lane = tid & 31;
  const int wave = tid >> 5;
  const int n    = blockIdx.z;
  const int M0   = blockIdx.y * 128;
  const int N0   = blockIdx.x * 128;
  const float* xn = x + (size_t)n * 256 * 4096;

  if (tid < 128) {
    int o = M0 + tid;
    float s = g1[o] * rsqrtf(v1[o] + BN_EPS);
    sc[tid] = s;
    sh[tid] = b1[o] - m1[o] * s;
  }

  const int wm = wave >> 2;
  const int wn = wave & 3;

  v8f acc[4][2];
#pragma unroll
  for (int mt = 0; mt < 4; ++mt)
#pragma unroll
    for (int nt = 0; nt < 2; ++nt)
#pragma unroll
      for (int r = 0; r < 8; ++r) acc[mt][nt][r] = 0.0f;

  for (int k0 = 0; k0 < 256; k0 += 32) {
    __syncthreads();
    {  // W tile 128x32
      int row = tid >> 1;
      int kk  = (tid & 1) * 16;
      const float* src = w + (size_t)(M0 + row) * 256 + k0 + kk;
#pragma unroll
      for (int u = 0; u < 16; ++u) Wl[row * 34 + kk + u] = (bf16)src[u];
    }
    {  // X tile 32x128, stored transposed [col][k]
      int k  = tid >> 3;
      int cb = (tid & 7) * 16;
      const float* src = xn + (size_t)(k0 + k) * 4096 + N0 + cb;
      if (k0 + 32 < 256)
        __builtin_prefetch(xn + (size_t)(k0 + 32 + k) * 4096 + N0 + cb, 0, 1);
#pragma unroll
      for (int u = 0; u < 16; ++u) Xl[(cb + u) * 34 + k] = (bf16)src[u];
    }
    __syncthreads();

    v16bf fa[4], fb[2];
#pragma unroll
    for (int mt = 0; mt < 4; ++mt)
      fa[mt] = load_frag16(&Wl[(wm * 64 + mt * 16) * 34], 34);
#pragma unroll
    for (int nt = 0; nt < 2; ++nt)
      fb[nt] = load_frag16(&Xl[(wn * 32 + nt * 16) * 34], 34);
#pragma unroll
    for (int mt = 0; mt < 4; ++mt)
#pragma unroll
      for (int nt = 0; nt < 2; ++nt)
        acc[mt][nt] = WMMA_BF16(fa[mt], fb[nt], acc[mt][nt]);
  }

  const int rbase = (lane < 16) ? 0 : 8;
  const int cl    = lane & 15;
#pragma unroll
  for (int mt = 0; mt < 4; ++mt)
#pragma unroll
    for (int nt = 0; nt < 2; ++nt)
#pragma unroll
      for (int r = 0; r < 8; ++r) {
        int ol  = wm * 64 + mt * 16 + rbase + r;
        int o   = M0 + ol;
        int col = N0 + wn * 32 + nt * 16 + cl;
        int h = col >> 6, wcol = col & 63;
        float vout = acc[mt][nt][r] * sc[ol] + sh[ol];
        qkv[((size_t)(n * 64 + wcol) * 512 + o) * 64 + h] = (bf16)vout;
      }
}

// ---------------------------------------------------------------------------
// Kernel 2: one workgroup per (b, head). All matmul-like work on WMMA:
//   qk (WMMA) -> logits; qr/kr as Rq=q^T*Pq / Rk=k^T*Pk WMMA with fused
//   banded ds_add_f32 scatter; softmax; sv = v*attn^T (WMMA);
//   svemb = Pv * skewed(attn)^T (WMMA). qkv tile arrives via
//   global_load_async_to_lds_b128 (ASYNCcnt).
// LDS arena (52KB) with lifetime aliasing.
// ---------------------------------------------------------------------------
#define SM_QT   0        // q^T [i][c] 64x32 bf16 (4KB)   | at aliases [0,8192)
#define SM_KT   4096     // k^T [j][c] 64x32 bf16 (4KB)
#define SM_VL   8192     // v [c][j] 32x64 bf16 (4KB)
#define SM_PV   12288    // Pv [c][d] 32x128 bf16 (8KB, d=127 zero)
#define SM_LG   20480    // logits f32 64x64 (16KB) | raw qkv aliases first 8KB
#define SM_PQ   36864    // Pq [d][c] 128x32 bf16 (8KB) | skA aliases [36864,53248)
#define SM_PK   45056    // Pk [d][c] 128x32 bf16 (8KB)
#define SM_TOT  53248

__global__ __launch_bounds__(256)
void axial_attn(const bf16* __restrict__ qkv, const float* __restrict__ pos_emb,
                const float* __restrict__ g2, const float* __restrict__ b2,
                const float* __restrict__ m2, const float* __restrict__ v2,
                const float* __restrict__ g3, const float* __restrict__ b3,
                const float* __restrict__ m3, const float* __restrict__ v3,
                float* __restrict__ yT) {
  __shared__ __align__(16) char smem[SM_TOT];
  bf16*  qT  = (bf16*)(smem + SM_QT);
  bf16*  kT  = (bf16*)(smem + SM_KT);
  bf16*  at  = (bf16*)(smem + SM_QT);   // alias: attn bf16 [i][j], after qT/kT die
  bf16*  vl  = (bf16*)(smem + SM_VL);
  bf16*  Pv  = (bf16*)(smem + SM_PV);
  float* lg  = (float*)(smem + SM_LG);
  bf16*  raw = (bf16*)(smem + SM_LG);   // alias: raw qkv block during load
  bf16*  Pq  = (bf16*)(smem + SM_PQ);
  bf16*  Pk  = (bf16*)(smem + SM_PK);
  bf16*  skA = (bf16*)(smem + SM_PQ);   // alias: skewed attn [i][d] 64x128

  const int tid  = threadIdx.x;
  const int lane = tid & 31;
  const int wave = tid >> 5;
  const int bid  = blockIdx.x;
  const int head = bid & 7;
  const int b    = bid >> 3;

  // --- Issue async copy of this (b,head)'s 8KB bf16 block into `raw` -------
  {
    const char* g = (const char*)(qkv + ((size_t)b * 512 + head * 64) * 64) + tid * 32;
    unsigned int l  = (unsigned int)(unsigned long long)(smem + SM_LG + tid * 32);
    unsigned long long ga = (unsigned long long)g;
    asm volatile("global_load_async_to_lds_b128 %0, %1, off\n\t"
                 "global_load_async_to_lds_b128 %0, %1, off offset:16"
                 :: "v"(l), "v"(ga) : "memory");
  }

  // --- Zero qT/kT (pad K 16..31), Pv (pad d=127), Pq/Pk (pads) -------------
  {
    const float4 z4 = make_float4(0.f, 0.f, 0.f, 0.f);
#pragma unroll
    for (int u = 0; u < 2; ++u) {             // [0,8192): qT,kT
      *(float4*)(smem + SM_QT + tid * 32 + u * 16) = z4;
      *(float4*)(smem + SM_PV + tid * 32 + u * 16) = z4;
    }
#pragma unroll
    for (int u = 0; u < 4; ++u)               // [36864,53248): Pq,Pk
      *(float4*)(smem + SM_PQ + tid * 64 + u * 16) = z4;
  }
  __syncthreads();

  // --- Fill Pq/Pk/Pv from pos_emb (fp32 -> bf16) ---------------------------
  {
    int r  = tid >> 2;
    int db = (tid & 3) * 32;
    const float* src = pos_emb + (size_t)r * 127;
    for (int u = 0; u < 32; ++u) {
      int d = db + u;
      if (d < 127) {
        bf16 val = (bf16)src[d];
        if (r < 16)      Pq[d * 32 + r] = val;
        else if (r < 32) Pk[d * 32 + (r - 16)] = val;
        else             Pv[(r - 32) * 128 + d] = val;
      }
    }
  }

  // --- Wait for async block, then transpose raw -> qT/kT/vl ----------------
  asm volatile("s_wait_asynccnt 0" ::: "memory");
  __syncthreads();
  {
    int ch = tid >> 2;
    int hb = (tid & 3) * 16;
#pragma unroll
    for (int u = 0; u < 16; ++u) {
      int h = hb + u;
      bf16 val = raw[ch * 64 + h];
      if (ch < 16)      qT[h * 32 + ch] = val;
      else if (ch < 32) kT[h * 32 + (ch - 16)] = val;
      else              vl[(ch - 32) * 64 + h] = val;
    }
  }
  __syncthreads();

  // --- BN2 folded constants ------------------------------------------------
  float s_qk = g2[head]      * rsqrtf(v2[head] + BN_EPS);
  float t_qk = b2[head]      - m2[head] * s_qk;
  float s_qr = g2[head + 8]  * rsqrtf(v2[head + 8] + BN_EPS);
  float t_qr = b2[head + 8]  - m2[head + 8] * s_qr;
  float s_kr = g2[head + 16] * rsqrtf(v2[head + 16] + BN_EPS);
  float t_kr = b2[head + 16] - m2[head + 16] * s_kr;
  const float tsum = t_qk + t_qr + t_kr;
  const int rb = (lane < 16) ? 0 : 8;
  const int cl = lane & 15;

  // --- qk = q^T k via WMMA (K=16 padded to 32); 16 tiles over 8 waves ------
  {
    int mt = wave >> 1;
    int nt = (wave & 1) * 2;
    v16bf aq = load_frag16(qT + mt * 16 * 32, 32);
#pragma unroll
    for (int t = 0; t < 2; ++t) {
      v16bf bk = load_frag16(kT + (nt + t) * 16 * 32, 32);
      v8f c;
#pragma unroll
      for (int r = 0; r < 8; ++r) c[r] = 0.0f;
      c = WMMA_BF16(aq, bk, c);
      int j = (nt + t) * 16 + cl;
#pragma unroll
      for (int r = 0; r < 8; ++r)
        lg[(mt * 16 + rb + r) * 64 + j] = s_qk * c[r] + tsum;
    }
  }
  __syncthreads();

  // --- qr/kr: Rq = q^T*Pq, Rk = k^T*Pk via WMMA with fused banded scatter --
  // qr[i,j] = Rq[i, i-j+63]  => element (i,d) -> j = i-d+63 (unique cell)
  // kr[i,j] = Rk[j, j-i+63]  => element (j,d) -> i = j-d+63 (unique cell)
  {
    int mt  = wave >> 1;        // 0..3 (rows of Rq/Rk)
    int ntb = (wave & 1) * 4;   // d-tile base: 0 or 4 (of 8)
    v16bf fq = load_frag16(qT + mt * 16 * 32, 32);
    v16bf fk = load_frag16(kT + mt * 16 * 32, 32);
#pragma unroll
    for (int nt = 0; nt < 4; ++nt) {
      int d = (ntb + nt) * 16 + cl;
      v16bf fpq = load_frag16(Pq + (ntb + nt) * 16 * 32, 32);
      v8f cq;
#pragma unroll
      for (int r = 0; r < 8; ++r) cq[r] = 0.0f;
      cq = WMMA_BF16(fq, fpq, cq);
#pragma unroll
      for (int r = 0; r < 8; ++r) {
        int i = mt * 16 + rb + r;
        int j = i - d + 63;
        if (j >= 0 && j < 64) atomicAdd(&lg[i * 64 + j], s_qr * cq[r]);
      }
      v16bf fpk = load_frag16(Pk + (ntb + nt) * 16 * 32, 32);
      v8f ck;
#pragma unroll
      for (int r = 0; r < 8; ++r) ck[r] = 0.0f;
      ck = WMMA_BF16(fk, fpk, ck);
#pragma unroll
      for (int r = 0; r < 8; ++r) {
        int j = mt * 16 + rb + r;
        int i = j - d + 63;
        if (i >= 0 && i < 64) atomicAdd(&lg[i * 64 + j], s_kr * ck[r]);
      }
    }
  }
  __syncthreads();

  // --- Softmax over j, one row per thread; write bf16 attn (aliases qT/kT) -
  if (tid < 64) {
    float* row = &lg[tid * 64];
    float mx = row[0];
    for (int j = 1; j < 64; ++j) mx = fmaxf(mx, row[j]);
    float s = 0.0f;
    for (int j = 0; j < 64; ++j) {
      float e = __expf(row[j] - mx);
      row[j] = e;
      s += e;
    }
    float inv = 1.0f / s;
    for (int j = 0; j < 64; ++j) at[tid * 64 + j] = (bf16)(row[j] * inv);
  }
  __syncthreads();

  // --- Build skewed attention skA[i][d] = attn[i, i-d+63] (aliases Pq/Pk) --
  {
    const float4 z4 = make_float4(0.f, 0.f, 0.f, 0.f);
#pragma unroll
    for (int u = 0; u < 4; ++u)
      *(float4*)(smem + SM_PQ + tid * 64 + u * 16) = z4;
  }
  __syncthreads();
  {
    int i  = tid >> 2;
    int jb = (tid & 3) * 16;
#pragma unroll
    for (int u = 0; u < 16; ++u) {
      int j = jb + u;
      skA[i * 128 + (i - j + 63)] = at[i * 64 + j];
    }
  }
  __syncthreads();

  // --- sv = v*attn^T (K=64) and svemb = Pv*skA^T (K=128) via WMMA ----------
  {
    int ct = wave >> 2;  // 0..1
    int it = wave & 3;   // 0..3
    v8f cs, ce;
#pragma unroll
    for (int r = 0; r < 8; ++r) { cs[r] = 0.0f; ce[r] = 0.0f; }
#pragma unroll
    for (int j0 = 0; j0 < 64; j0 += 32) {
      v16bf fa = load_frag16(vl + ct * 16 * 64 + j0, 64);
      v16bf fb = load_frag16(at + it * 16 * 64 + j0, 64);
      cs = WMMA_BF16(fa, fb, cs);
    }
#pragma unroll
    for (int d0 = 0; d0 < 128; d0 += 32) {
      v16bf fa = load_frag16(Pv + ct * 16 * 128 + d0, 128);
      v16bf fb = load_frag16(skA + it * 16 * 128 + d0, 128);
      ce = WMMA_BF16(fa, fb, ce);
    }
    int i = it * 16 + cl;
#pragma unroll
    for (int r = 0; r < 8; ++r) {
      int cc = ct * 16 + rb + r;
      lg[cc * 64 + i] = cs[r];          // sv staging   (lg reused)
      lg[2048 + cc * 64 + i] = ce[r];   // svemb staging
    }
  }
  __syncthreads();

  // --- BN3 + paired sum; coalesced store of yT[b][head*32+c][i] ------------
  {
    int c  = tid >> 3;
    int ib = (tid & 7) * 8;
    int ch = head * 64 + 2 * c;
    float s3a = g3[ch] * rsqrtf(v3[ch] + BN_EPS);
    float t3a = b3[ch] - m3[ch] * s3a;
    float s3b = g3[ch + 1] * rsqrtf(v3[ch + 1] + BN_EPS);
    float t3b = b3[ch + 1] - m3[ch + 1] * s3b;
    float* dst = yT + ((size_t)b * 256 + head * 32 + c) * 64;
#pragma unroll
    for (int u = 0; u < 8; ++u) {
      int i = ib + u;
      dst[i] = s3a * lg[c * 64 + i] + t3a + s3b * lg[2048 + c * 64 + i] + t3b;
    }
  }
}

// ---------------------------------------------------------------------------
// Kernel 3: transpose yT[(n*64+w)][o][h] -> out[n][o][h][w] (LDS tiled).
// ---------------------------------------------------------------------------
__global__ __launch_bounds__(256)
void transpose_out(const float* __restrict__ yT, float* __restrict__ out) {
  __shared__ float tile[64][65];
  const int o   = blockIdx.x;
  const int n   = blockIdx.y;
  const int tid = threadIdx.x;
  {
    int w  = tid >> 2;
    int hb = (tid & 3) * 16;
    const float* src = yT + ((size_t)(n * 64 + w) * 256 + o) * 64 + hb;
#pragma unroll
    for (int u = 0; u < 16; ++u) tile[w][hb + u] = src[u];
  }
  __syncthreads();
  {
    int h  = tid >> 2;
    int wb = (tid & 3) * 16;
    float* dst = out + ((size_t)(n * 256 + o) * 64 + h) * 64 + wb;
#pragma unroll
    for (int u = 0; u < 16; ++u) dst[u] = tile[wb + u][h];
  }
}

extern "C" void kernel_launch(void* const* d_in, const int* in_sizes, int n_in,
                              void* d_out, int out_size, void* d_ws, size_t ws_size,
                              hipStream_t stream) {
  const float* x  = (const float*)d_in[0];
  const float* wq = (const float*)d_in[1];
  const float* g1 = (const float*)d_in[2];
  const float* b1 = (const float*)d_in[3];
  const float* m1 = (const float*)d_in[4];
  const float* v1 = (const float*)d_in[5];
  const float* pe = (const float*)d_in[6];
  const float* g2 = (const float*)d_in[7];
  const float* b2 = (const float*)d_in[8];
  const float* m2 = (const float*)d_in[9];
  const float* v2 = (const float*)d_in[10];
  const float* g3 = (const float*)d_in[11];
  const float* b3 = (const float*)d_in[12];
  const float* m3 = (const float*)d_in[13];
  const float* v3 = (const float*)d_in[14];
  float* out = (float*)d_out;

  bf16*  qkv = (bf16*)d_ws;
  float* yT  = (float*)((char*)d_ws + (size_t)1024 * 512 * 64 * 2);

  qkv_gemm_bn<<<dim3(32, 4, 16), 256, 0, stream>>>(x, wq, g1, b1, m1, v1, qkv);
  axial_attn<<<dim3(8192), 256, 0, stream>>>(qkv, pe, g2, b2, m2, v2,
                                             g3, b3, m3, v3, yT);
  transpose_out<<<dim3(256, 16), 256, 0, stream>>>(yT, out);
}
